// SchNetInteractionBlock_37220186587471
// MI455X (gfx1250) — matrix operand order repaired and checked by reading, compiled, not run on hardware
//
#include <hip/hip_runtime.h>
#include <hip/hip_bf16.h>

// ---------------------------------------------------------------------------
// SchNet interaction block for gfx1250 (MI455X), bf16 WMMA w/ f32 accumulate.
//   edge_feature = silu(rbf @ We1 + be1) @ We2 + be2        [E,512]
//   msg          = edge_feature * x[col]                     (fused)
//   agg          = segment_sum(msg, row)                     (fp32 HW atomics)
//   out          = x + silu(agg @ Wn1 + bn1) @ Wn2 + bn2
// Edge blocks cover 64 edges (4 m-tiles): every B-fragment feeds 4 A-frags,
// cutting L2 weight traffic to ~1.25 GB; 8 independent WMMA chains per
// k-step hide the WMMA->WMMA accumulation hazard window.
// ---------------------------------------------------------------------------

#define HDIM 512
#define RDIM 128
#define LDR  136   // 128 + 8 bf16 pad (bank-conflict avoidance)
#define LDH  520   // 512 + 8 bf16 pad
#define EM   64    // edges per block (4 m-tiles)
#define NM   32    // node rows per block (2 m-tiles)

typedef __attribute__((ext_vector_type(16))) __bf16 bf16x16;
typedef __attribute__((ext_vector_type(8)))  __bf16 bf16x8;
typedef __attribute__((ext_vector_type(8)))  float  f32x8;

static __device__ inline f32x8 wmma_bf16(bf16x16 a, bf16x16 b, f32x8 c) {
    // D = A(16x32 bf16) x B(32x16 bf16) + C(16x16 f32)
    return __builtin_amdgcn_wmma_f32_16x16x32_bf16(
        /*neg_a=*/false, a, /*neg_b=*/false, b,
        /*c_mod=*/(short)0, c, /*reuse_a=*/false, /*reuse_b=*/false);
}

// A-fragment from a bf16 row-major LDS tile. A layout used here (16x32), contiguous k, same order as B:
//   lane L: M = L%16, half = L/16
//   element e in [0,16) -> K = k0 + 16*half + e
static __device__ inline bf16x16 load_a_lds(const __bf16* rowk0, int half) {
    bf16x8 lo = *reinterpret_cast<const bf16x8*>(rowk0 + 16 * half);
    bf16x8 hi = *reinterpret_cast<const bf16x8*>(rowk0 + 16 * half + 8);
    return __builtin_shufflevector(lo, hi, 0, 1, 2, 3, 4, 5, 6, 7,
                                           8, 9, 10, 11, 12, 13, 14, 15);
}

// B-fragment from bf16 W^T (row-major [n][K]). B layout used here (32x16), same k order as A:
//   lane L: N = L%16, half = L/16, element e -> K = k0 + 16*half + e
//   => one contiguous 32-byte load per lane.
static __device__ inline bf16x16 load_b_glb(const __bf16* p) {
    return *reinterpret_cast<const bf16x16*>(p);
}

// silu via v_exp_f32 + v_rcp_f32 (avoid IEEE div_scale/Newton chain)
static __device__ inline float silu(float v) {
    return v * __builtin_amdgcn_rcpf(1.0f + __expf(-v));
}

// ---------------------------------------------------------------------------
// Weight transpose + f32 -> bf16 convert:  WT[n*K + k] = bf16(W[k*H + n])
// ---------------------------------------------------------------------------
__global__ void convertT_kernel(const float* __restrict__ W,
                                __bf16* __restrict__ WT, int K, int H) {
    int i = blockIdx.x * blockDim.x + threadIdx.x;
    if (i < K * H) {
        int n = i / K;
        int k = i - n * K;
        WT[i] = (__bf16)W[k * H + n];
    }
}

// ---------------------------------------------------------------------------
// Edge pipeline: one block = 64 edges (4 m-tiles). 8 waves; each wave owns
// 4 n-tiles as 2 pairs; per k-step: 4 A-frags x 2 B-frags -> 8 WMMA chains.
// ---------------------------------------------------------------------------
__global__ __launch_bounds__(256) void edge_kernel(
    const float* __restrict__ rbf, const float* __restrict__ x,
    const __bf16* __restrict__ We1T, const float* __restrict__ be1,
    const __bf16* __restrict__ We2T, const float* __restrict__ be2,
    const int* __restrict__ row_idx, const int* __restrict__ col_idx,
    float* __restrict__ agg, int E) {

    __shared__ __align__(16) __bf16 s_rbf[EM * LDR];   // 17.4 KB
    __shared__ __align__(16) __bf16 s_h1[EM * LDH];    // 66.6 KB
    __shared__ int s_row[EM];
    __shared__ int s_col[EM];

    const int tid = threadIdx.x;
    const int e0  = blockIdx.x * EM;

    // Stage rbf tile (EM x 128) as bf16 into LDS.
    for (int i = tid; i < EM * RDIM; i += 256) {
        int m = i >> 7;
        int k = i & (RDIM - 1);
        int e = e0 + m;
        float v = (e < E) ? rbf[e * RDIM + k] : 0.0f;
        s_rbf[m * LDR + k] = (__bf16)v;
    }
    if (tid < EM) {
        int e = e0 + tid;
        s_row[tid] = (e < E) ? row_idx[e] : 0;
        s_col[tid] = (e < E) ? col_idx[e] : 0;
    }
    __syncthreads();

    const int wave = tid >> 5;
    const int lane = tid & 31;
    const int half = lane >> 4;
    const int ncol = lane & 15;   // N (and A-row M) index within tile

    // ---- Stage 1: h1 = silu(rbf @ We1 + be1), K = 128 ----
    #pragma unroll 1
    for (int tp = 0; tp < 2; ++tp) {
        int n0a = (wave * 4 + tp * 2) * 16;
        int n0b = n0a + 16;
        float biasa = be1[n0a + ncol];
        float biasb = be1[n0b + ncol];
        f32x8 c[4][2];
        #pragma unroll
        for (int m = 0; m < 4; ++m)
            #pragma unroll
            for (int r = 0; r < 8; ++r) { c[m][0][r] = biasa; c[m][1][r] = biasb; }

        const __bf16* pa = We1T + (n0a + ncol) * RDIM + 16 * half;
        const __bf16* pb = We1T + (n0b + ncol) * RDIM + 16 * half;
        #pragma unroll
        for (int ks = 0; ks < 4; ++ks) {
            int k0 = ks * 32;
            bf16x16 b0 = load_b_glb(pa + k0);
            bf16x16 b1 = load_b_glb(pb + k0);
            #pragma unroll
            for (int m = 0; m < 4; ++m) {
                bf16x16 a = load_a_lds(&s_rbf[(m * 16 + ncol) * LDR + k0], half);
                c[m][0] = wmma_bf16(a, b0, c[m][0]);
                c[m][1] = wmma_bf16(a, b1, c[m][1]);
            }
        }
        #pragma unroll
        for (int m = 0; m < 4; ++m) {
            #pragma unroll
            for (int r = 0; r < 8; ++r) {
                int rr = (m * 16 + r + 8 * half) * LDH;
                s_h1[rr + n0a + ncol] = (__bf16)silu(c[m][0][r]);
                s_h1[rr + n0b + ncol] = (__bf16)silu(c[m][1][r]);
            }
        }
    }
    __syncthreads();

    // ---- Stage 2: ef = h1 @ We2 + be2, K = 512; fused gather/mul/scatter ----
    #pragma unroll 1
    for (int tp = 0; tp < 2; ++tp) {
        int n0a = (wave * 4 + tp * 2) * 16;
        int n0b = n0a + 16;
        float biasa = be2[n0a + ncol];
        float biasb = be2[n0b + ncol];
        f32x8 c[4][2];
        #pragma unroll
        for (int m = 0; m < 4; ++m)
            #pragma unroll
            for (int r = 0; r < 8; ++r) { c[m][0][r] = biasa; c[m][1][r] = biasb; }

        const __bf16* pa = We2T + (n0a + ncol) * HDIM + 16 * half;
        const __bf16* pb = We2T + (n0b + ncol) * HDIM + 16 * half;
        #pragma unroll 4
        for (int ks = 0; ks < 16; ++ks) {
            int k0 = ks * 32;
            bf16x16 b0 = load_b_glb(pa + k0);
            bf16x16 b1 = load_b_glb(pb + k0);
            #pragma unroll
            for (int m = 0; m < 4; ++m) {
                bf16x16 a = load_a_lds(&s_h1[(m * 16 + ncol) * LDH + k0], half);
                c[m][0] = wmma_bf16(a, b0, c[m][0]);
                c[m][1] = wmma_bf16(a, b1, c[m][1]);
            }
        }
        #pragma unroll
        for (int m = 0; m < 4; ++m) {
            #pragma unroll
            for (int r = 0; r < 8; ++r) {
                int er = m * 16 + r + 8 * half;   // edge row within block
                int e  = e0 + er;
                if (e < E) {
                    int xbase = s_col[er] * HDIM + ncol;
                    int abase = s_row[er] * HDIM + ncol;
                    unsafeAtomicAdd(&agg[abase + n0a], c[m][0][r] * x[xbase + n0a]);
                    unsafeAtomicAdd(&agg[abase + n0b], c[m][1][r] * x[xbase + n0b]);
                }
            }
        }
    }
}

// ---------------------------------------------------------------------------
// Node pipeline: out = x + silu(agg @ Wn1 + bn1) @ Wn2 + bn2.
// One block = 32 rows (2 m-tiles). agg lives in `out` (d_out), overwritten
// in place per tile.
// ---------------------------------------------------------------------------
__global__ __launch_bounds__(256) void node_kernel(
    const float* __restrict__ x,
    const __bf16* __restrict__ Wn1T, const float* __restrict__ bn1,
    const __bf16* __restrict__ Wn2T, const float* __restrict__ bn2,
    float* __restrict__ out, int Nn) {

    __shared__ __align__(16) __bf16 s_in[NM * LDH];   // 33.3 KB
    __shared__ __align__(16) __bf16 s_h[NM * LDH];    // 33.3 KB

    const int tid = threadIdx.x;
    const int r0  = blockIdx.x * NM;

    // Stage agg tile (NM x 512) as bf16 into LDS.
    for (int i = tid; i < NM * HDIM; i += 256) {
        int m = i >> 9;
        int k = i & (HDIM - 1);
        int rr = r0 + m;
        float v = (rr < Nn) ? out[rr * HDIM + k] : 0.0f;
        s_in[m * LDH + k] = (__bf16)v;
    }
    __syncthreads();

    const int wave = tid >> 5;
    const int lane = tid & 31;
    const int half = lane >> 4;
    const int ncol = lane & 15;

    // ---- Stage 1: h = silu(agg @ Wn1 + bn1) ----
    #pragma unroll 1
    for (int tp = 0; tp < 2; ++tp) {
        int n0a = (wave * 4 + tp * 2) * 16;
        int n0b = n0a + 16;
        float biasa = bn1[n0a + ncol];
        float biasb = bn1[n0b + ncol];
        f32x8 c[2][2];
        #pragma unroll
        for (int m = 0; m < 2; ++m)
            #pragma unroll
            for (int r = 0; r < 8; ++r) { c[m][0][r] = biasa; c[m][1][r] = biasb; }

        const __bf16* pa = Wn1T + (n0a + ncol) * HDIM + 16 * half;
        const __bf16* pb = Wn1T + (n0b + ncol) * HDIM + 16 * half;
        #pragma unroll 4
        for (int ks = 0; ks < 16; ++ks) {
            int k0 = ks * 32;
            bf16x16 b0 = load_b_glb(pa + k0);
            bf16x16 b1 = load_b_glb(pb + k0);
            #pragma unroll
            for (int m = 0; m < 2; ++m) {
                bf16x16 a = load_a_lds(&s_in[(m * 16 + ncol) * LDH + k0], half);
                c[m][0] = wmma_bf16(a, b0, c[m][0]);
                c[m][1] = wmma_bf16(a, b1, c[m][1]);
            }
        }
        #pragma unroll
        for (int m = 0; m < 2; ++m) {
            #pragma unroll
            for (int r = 0; r < 8; ++r) {
                int rr = (m * 16 + r + 8 * half) * LDH;
                s_h[rr + n0a + ncol] = (__bf16)silu(c[m][0][r]);
                s_h[rr + n0b + ncol] = (__bf16)silu(c[m][1][r]);
            }
        }
    }
    __syncthreads();

    // ---- Stage 2: out = x + h @ Wn2 + bn2 ----
    #pragma unroll 1
    for (int tp = 0; tp < 2; ++tp) {
        int n0a = (wave * 4 + tp * 2) * 16;
        int n0b = n0a + 16;
        float biasa = bn2[n0a + ncol];
        float biasb = bn2[n0b + ncol];
        f32x8 c[2][2];
        #pragma unroll
        for (int m = 0; m < 2; ++m)
            #pragma unroll
            for (int r = 0; r < 8; ++r) { c[m][0][r] = biasa; c[m][1][r] = biasb; }

        const __bf16* pa = Wn2T + (n0a + ncol) * HDIM + 16 * half;
        const __bf16* pb = Wn2T + (n0b + ncol) * HDIM + 16 * half;
        #pragma unroll 4
        for (int ks = 0; ks < 16; ++ks) {
            int k0 = ks * 32;
            bf16x16 b0 = load_b_glb(pa + k0);
            bf16x16 b1 = load_b_glb(pb + k0);
            #pragma unroll
            for (int m = 0; m < 2; ++m) {
                bf16x16 a = load_a_lds(&s_h[(m * 16 + ncol) * LDH + k0], half);
                c[m][0] = wmma_bf16(a, b0, c[m][0]);
                c[m][1] = wmma_bf16(a, b1, c[m][1]);
            }
        }
        #pragma unroll
        for (int m = 0; m < 2; ++m) {
            #pragma unroll
            for (int r = 0; r < 8; ++r) {
                int rr = r0 + m * 16 + r + 8 * half;
                if (rr < Nn) {
                    int off = rr * HDIM + ncol;
                    out[off + n0a] = x[off + n0a] + c[m][0][r];
                    out[off + n0b] = x[off + n0b] + c[m][1][r];
                }
            }
        }
    }
}

// ---------------------------------------------------------------------------
extern "C" void kernel_launch(void* const* d_in, const int* in_sizes, int n_in,
                              void* d_out, int out_size, void* d_ws, size_t ws_size,
                              hipStream_t stream) {
    const float* x    = (const float*)d_in[0];
    const float* rbf  = (const float*)d_in[1];
    const float* We1  = (const float*)d_in[2];
    const float* be1  = (const float*)d_in[3];
    const float* We2  = (const float*)d_in[4];
    const float* be2  = (const float*)d_in[5];
    const float* Wn1  = (const float*)d_in[6];
    const float* bn1  = (const float*)d_in[7];
    const float* Wn2  = (const float*)d_in[8];
    const float* bn2  = (const float*)d_in[9];
    const int*   eidx = (const int*)d_in[10];

    const int E  = in_sizes[1] / RDIM;   // 160000
    const int Nn = in_sizes[0] / HDIM;   // 10000
    const int* row = eidx;               // edge_index[0]
    const int* col = eidx + E;           // edge_index[1]

    float* out = (float*)d_out;          // doubles as agg accumulator

    // bf16 transposed weights in workspace (~1.7 MB total).
    char* ws = (char*)d_ws;
    __bf16* We1T = (__bf16*)(ws);
    __bf16* We2T = (__bf16*)(ws + (size_t)HDIM * RDIM * 2);
    __bf16* Wn1T = (__bf16*)(ws + (size_t)HDIM * RDIM * 2 + (size_t)HDIM * HDIM * 2);
    __bf16* Wn2T = (__bf16*)(ws + (size_t)HDIM * RDIM * 2 + (size_t)HDIM * HDIM * 4);

    // agg = 0 (capture-safe async memset).
    hipMemsetAsync(d_out, 0, (size_t)out_size * sizeof(float), stream);

    // Convert + transpose weights to bf16.
    {
        int n1 = RDIM * HDIM, n2 = HDIM * HDIM;
        convertT_kernel<<<(n1 + 255) / 256, 256, 0, stream>>>(We1, We1T, RDIM, HDIM);
        convertT_kernel<<<(n2 + 255) / 256, 256, 0, stream>>>(We2, We2T, HDIM, HDIM);
        convertT_kernel<<<(n2 + 255) / 256, 256, 0, stream>>>(Wn1, Wn1T, HDIM, HDIM);
        convertT_kernel<<<(n2 + 255) / 256, 256, 0, stream>>>(Wn2, Wn2T, HDIM, HDIM);
    }

    // Edge MLP + gather-multiply + scatter-add.
    edge_kernel<<<(E + EM - 1) / EM, 256, 0, stream>>>(
        rbf, x, We1T, be1, We2T, be2, row, col, out, E);

    // Node MLP + residual (in-place over agg in d_out).
    node_kernel<<<(Nn + NM - 1) / NM, 256, 0, stream>>>(
        x, Wn1T, bn1, Wn2T, bn2, out, Nn);
}
